// ParallelWarper_47863115546790
// MI455X (gfx1250) — compile-verified, matrix operand order checked by reading, not run on hardware
//
#include <hip/hip_runtime.h>
#include <hip/hip_bf16.h>
#include <math.h>

// ---------------------------------------------------------------------------
// BasicVSR++-style deformable alignment for MI455X (gfx1250, wave32, WMMA).
// Convs: implicit GEMM, register-blocked 16px x 64cout per wave, weights
// staged per-tap in LDS. DCN: LDS-staged modulated samples + WMMA over K=576.
// ---------------------------------------------------------------------------

typedef __attribute__((ext_vector_type(16))) _Float16     v16h;
typedef __attribute__((ext_vector_type(8)))  float        v8f;
typedef __attribute__((ext_vector_type(4)))  unsigned int v4u;

#define H_     128
#define W_     128
#define HW_    16384
#define C_     64
#define T_     6
#define NPAIR  10          // B * 2 * (T-1)
#define DG_    16
#define KPOS   9
#define CG_    4
#define CIN1   130         // 2C + 2
#define CIN1P  160         // padded to multiple of 32
#define C4OUT  432         // 3*K*DG
#define C4OUTP 448         // padded to multiple of 64
#define DCNK   576         // DG*CG*KPOS, = 18*32 exactly

// ---------------------------------------------------------------------------
// Stage 0a: flow-warp xi by flow (bilinear, zero pad) and build f16 NHWC
// input tensor [N][H*W][CIN1P]: {warped(64), cur(64), flow(2), zeros(30)}
// ---------------------------------------------------------------------------
__global__ __launch_bounds__(256)
void warp_build_inp(const float* __restrict__ x,
                    const float* __restrict__ flb,
                    const float* __restrict__ flf,
                    _Float16* __restrict__ inp)
{
    int gid = blockIdx.x * blockDim.x + threadIdx.x;
    if (gid >= NPAIR * HW_) return;
    int n = gid / HW_, p = gid - n * HW_;
    int y = p >> 7, xx = p & 127;

    const float* xi = x + (size_t)((n < 5) ? (n + 1) : (n - 5)) * C_ * HW_;
    const float* cu = x + (size_t)((n < 5) ?  n      : (n - 4)) * C_ * HW_;
    const float* fl = (n < 5) ? (flb + (size_t)n * 2 * HW_)
                              : (flf + (size_t)(n - 5) * 2 * HW_);

    float fx = fl[p];            // channel 0 = x-displacement
    float fy = fl[HW_ + p];      // channel 1 = y-displacement
    float py = (float)y + fy;
    float px = (float)xx + fx;
    float y0f = floorf(py), x0f = floorf(px);
    int   y0 = (int)y0f,    x0 = (int)x0f;
    float wy = py - y0f,    wx = px - x0f;

    bool vy0 = (unsigned)y0       < (unsigned)H_;
    bool vy1 = (unsigned)(y0 + 1) < (unsigned)H_;
    bool vx0 = (unsigned)x0       < (unsigned)W_;
    bool vx1 = (unsigned)(x0 + 1) < (unsigned)W_;
    int cy0 = min(max(y0, 0), H_ - 1),     cy1 = min(max(y0 + 1, 0), H_ - 1);
    int cx0 = min(max(x0, 0), W_ - 1),     cx1 = min(max(x0 + 1, 0), W_ - 1);
    float w00 = (1.f - wy) * (1.f - wx) * ((vy0 && vx0) ? 1.f : 0.f);
    float w01 = (1.f - wy) * wx         * ((vy0 && vx1) ? 1.f : 0.f);
    float w10 = wy * (1.f - wx)         * ((vy1 && vx0) ? 1.f : 0.f);
    float w11 = wy * wx                 * ((vy1 && vx1) ? 1.f : 0.f);
    int i00 = cy0 * W_ + cx0, i01 = cy0 * W_ + cx1;
    int i10 = cy1 * W_ + cx0, i11 = cy1 * W_ + cx1;

    _Float16* o = inp + (size_t)gid * CIN1P;
#pragma unroll 4
    for (int c = 0; c < C_; ++c) {
        const float* ch = xi + (size_t)c * HW_;
        float v = w00 * ch[i00] + w01 * ch[i01] + w10 * ch[i10] + w11 * ch[i11];
        o[c] = (_Float16)v;
    }
#pragma unroll 4
    for (int c = 0; c < C_; ++c)
        o[C_ + c] = (_Float16)cu[(size_t)c * HW_ + p];
    o[128] = (_Float16)fx;
    o[129] = (_Float16)fy;
#pragma unroll
    for (int c = CIN1; c < CIN1P; ++c) o[c] = (_Float16)0.f;
}

// ---------------------------------------------------------------------------
// Stage 0b: pre-swizzle conv weights (OIHW fp32) into WMMA B-matrix layout,
// grouped so each (64-cout block, tap) slice is contiguous for LDS staging:
//   wbuf[g64][tap][nt4][kc][lane][i] ; cout = (g64*4+nt4)*16 + lane%16,
//   K = kc*32 + (lane>>4)*8 + (i<8 ? i : i+8)   (16-bit B 32x16 lane layout)
// ---------------------------------------------------------------------------
__global__ __launch_bounds__(256)
void prep_conv_w(const float* __restrict__ w, _Float16* __restrict__ wbuf,
                 int COUT, int COUTP, int CIN, int CINP)
{
    int  KC = CINP / 32;
    long total = (long)(COUTP / 16) * KPOS * KC * 32 * 16;
    long gid = (long)blockIdx.x * blockDim.x + threadIdx.x;
    if (gid >= total) return;
    int i    = (int)(gid & 15);
    long r   = gid >> 4;
    int lane = (int)(r & 31);  r >>= 5;
    int kc   = (int)(r % KC);  r /= KC;
    int nt4  = (int)(r % 4);   r /= 4;
    int tap  = (int)(r % KPOS); r /= KPOS;
    int g64  = (int)r;
    int cout = (g64 * 4 + nt4) * 16 + (lane & 15);
    int kof  = ((lane >> 4) * 8) + ((i < 8) ? i : i + 8);
    int cin  = kc * 32 + kof;
    float v = 0.f;
    if (cout < COUT && cin < CIN)
        v = w[((size_t)cout * CIN + cin) * 9 + tap];
    wbuf[gid] = (_Float16)v;
}

// DCN weights: wr[o][g][c][k] -> B layout over combined K = (g*CG+c)*9 + k.
// Layout: [ntile(4)][kc(18)][lane][i]
__global__ __launch_bounds__(256)
void prep_dcn_w(const float* __restrict__ w, _Float16* __restrict__ wbuf)
{
    int total = 4 * 18 * 32 * 16;
    int gid = blockIdx.x * blockDim.x + threadIdx.x;
    if (gid >= total) return;
    int i    = gid & 15;
    int r    = gid >> 4;
    int lane = r & 31;  r >>= 5;
    int kc   = r % 18;  r /= 18;
    int nt   = r;
    int cout = nt * 16 + (lane & 15);
    int kof  = ((lane >> 4) * 8) + ((i < 8) ? i : i + 8);
    int kg   = kc * 32 + kof;        // < 576 always
    int ci   = kg / 9, k = kg - ci * 9;
    wbuf[gid] = (_Float16)w[((size_t)cout * C_ + ci) * 9 + k];
}

// ---------------------------------------------------------------------------
// 3x3 conv as implicit GEMM on WMMA.
// Block = 256 threads = 8 waves, tile = 128 pixels x 64 couts.
// Each wave: 16 pixels x 64 couts via 4 accumulators, reusing one A fragment
// per 4 WMMAs. Per tap, the 64-cout weight slice (4*KC*512 halves <= 20 KB)
// is cooperatively staged into LDS and consumed as ds_load_b128 B fragments.
// ---------------------------------------------------------------------------
template <int CINP, bool LEAKY>
__global__ __launch_bounds__(256)
void conv3x3_wmma(const _Float16* __restrict__ in,
                  const _Float16* __restrict__ wbuf,
                  const float*    __restrict__ bias,
                  _Float16*       __restrict__ out,
                  int COUT, int COUTP)
{
    constexpr int KC = CINP / 32;
    __shared__ __align__(16) _Float16 wsh[4 * KC * 512];

    int tid  = threadIdx.x;
    int lane = tid & 31;
    int wid  = tid >> 5;
    int n    = blockIdx.z;
    int pb   = blockIdx.x * 128 + wid * 16;   // this wave's 16-pixel M tile

    int prow = pb + (lane & 15);              // A-fragment row pixel
    int ry = prow >> 7, rx = prow & 127;
    int klo = (lane >> 4) * 8;                // lane's K sub-offset (A layout)

    const _Float16* ibase  = in + (size_t)n * HW_ * CINP;
    const _Float16* wslice = wbuf + (size_t)blockIdx.y * KPOS * 4 * KC * 512;

    v8f acc[4] = {{}, {}, {}, {}};
    v4u zero4 = {0u, 0u, 0u, 0u};

    for (int tap = 0; tap < KPOS; ++tap) {
        __syncthreads();   // protect LDS from previous tap's readers
        {
            const v4u* src = (const v4u*)(wslice + (size_t)tap * 4 * KC * 512);
            v4u*       dst = (v4u*)wsh;
#pragma unroll
            for (int j = 0; j < KC; ++j)      // 256*KC v4u total, KC per thread
                dst[tid + j * 256] = src[tid + j * 256];
        }
        __syncthreads();

        int sy = ry + (tap / 3) - 1;
        int sx = rx + (tap % 3) - 1;
        bool valid = ((unsigned)sy < (unsigned)H_) && ((unsigned)sx < (unsigned)W_);
        const _Float16* arow = ibase + (size_t)(sy * W_ + sx) * CINP + klo;

#pragma unroll
        for (int kc = 0; kc < KC; ++kc) {
            union { v4u u[2]; v16h h; } A;
            if (valid) {
                A.u[0] = *(const v4u*)(arow + kc * 32);
                A.u[1] = *(const v4u*)(arow + kc * 32 + 16);
            } else {
                A.u[0] = zero4; A.u[1] = zero4;
            }
#pragma unroll
            for (int nt = 0; nt < 4; ++nt) {
                union { v4u u[2]; v16h h; } B;
                const _Float16* bp = wsh + ((size_t)(nt * KC + kc) * 32 + lane) * 16;
                B.u[0] = *(const v4u*)(bp);
                B.u[1] = *(const v4u*)(bp + 8);
                acc[nt] = __builtin_amdgcn_wmma_f32_16x16x32_f16(
                    false, A.h, false, B.h, (short)0, acc[nt], false, false);
            }
        }
    }

    int pr0 = pb + (lane >> 4) * 8;           // D layout: row = 8*(lane>=16)+r
#pragma unroll
    for (int nt = 0; nt < 4; ++nt) {
        int cout = blockIdx.y * 64 + nt * 16 + (lane & 15);
        float bb = (cout < COUT) ? bias[cout] : 0.f;
        _Float16* ob = out + (size_t)n * HW_ * COUTP + cout;
#pragma unroll
        for (int r = 0; r < 8; ++r) {
            float v = acc[nt][r] + bb;
            if (LEAKY) v = (v >= 0.f) ? v : 0.1f * v;
            if (cout >= COUT) v = 0.f;
            ob[(size_t)(pr0 + r) * COUTP] = (_Float16)v;
        }
    }
}

// ---------------------------------------------------------------------------
// Modulated deformable conv. Per 32-pixel tile:
//   stage 1: 256 threads compute tanh/sigmoid offsets+mask from raw conv4 out,
//            bilinear-sample xi, modulate, stage s[32][576] f16 into LDS
//   stage 2: 8 waves (2M x 4N) run 18 WMMA K-chunks vs pre-swizzled weights,
//            add bias, scatter fp32 into x_backward / x_forward halves of out.
// ---------------------------------------------------------------------------
#define SROW 584   // 576 + 8 pad halves; 1168 B row keeps 16B alignment

__global__ __launch_bounds__(256)
void dcn_wmma(const float* __restrict__ x,
              const float* __restrict__ flb,
              const float* __restrict__ flf,
              const _Float16* __restrict__ c4,     // [N][HW][C4OUTP] raw
              const _Float16* __restrict__ wdbuf,
              const float*    __restrict__ bias,
              float*          __restrict__ out)
{
    __shared__ __align__(16) _Float16 smem[32 * SROW];
    int tid = threadIdx.x;
    int n   = blockIdx.z;
    int pb  = blockIdx.x * 32;

    const float* xi = x + (size_t)((n < 5) ? (n + 1) : (n - 5)) * C_ * HW_;
    const float* fl = (n < 5) ? (flb + (size_t)n * 2 * HW_)
                              : (flf + (size_t)(n - 5) * 2 * HW_);

    // ---- stage 1: sample + modulate into LDS -----------------------------
    for (int task = tid; task < 32 * DG_ * KPOS; task += 256) {
        int pl  = task / (DG_ * KPOS);
        int rem = task - pl * (DG_ * KPOS);
        int g = rem / KPOS, k = rem - g * KPOS;
        int p = pb + pl;
        int y = p >> 7, xx = p & 127;

        const _Float16* cp = c4 + ((size_t)n * HW_ + p) * C4OUTP;
        int ch = g * KPOS + k;
        float roy = (float)cp[2 * ch];
        float rox = (float)cp[2 * ch + 1];
        float rm  = (float)cp[288 + ch];
        float fy  = fl[HW_ + p], fx = fl[p];
        float oy  = 10.f * tanhf(roy) + fy;     // MRM * tanh + flipped flow
        float ox  = 10.f * tanhf(rox) + fx;
        float m   = 1.f / (1.f + __expf(-rm));  // sigmoid mask

        float py = (float)(y - 1 + k / 3) + oy;
        float px = (float)(xx - 1 + k % 3) + ox;
        float y0f = floorf(py), x0f = floorf(px);
        int   y0 = (int)y0f,    x0 = (int)x0f;
        float wy = py - y0f,    wx = px - x0f;
        bool vy0 = (unsigned)y0       < (unsigned)H_;
        bool vy1 = (unsigned)(y0 + 1) < (unsigned)H_;
        bool vx0 = (unsigned)x0       < (unsigned)W_;
        bool vx1 = (unsigned)(x0 + 1) < (unsigned)W_;
        int cy0 = min(max(y0, 0), H_ - 1), cy1 = min(max(y0 + 1, 0), H_ - 1);
        int cx0 = min(max(x0, 0), W_ - 1), cx1 = min(max(x0 + 1, 0), W_ - 1);
        float w00 = (1.f - wy) * (1.f - wx) * ((vy0 && vx0) ? 1.f : 0.f);
        float w01 = (1.f - wy) * wx         * ((vy0 && vx1) ? 1.f : 0.f);
        float w10 = wy * (1.f - wx)         * ((vy1 && vx0) ? 1.f : 0.f);
        float w11 = wy * wx                 * ((vy1 && vx1) ? 1.f : 0.f);
        int i00 = cy0 * W_ + cx0, i01 = cy0 * W_ + cx1;
        int i10 = cy1 * W_ + cx0, i11 = cy1 * W_ + cx1;

        const float* xg = xi + (size_t)(g * CG_) * HW_;
#pragma unroll
        for (int c = 0; c < CG_; ++c) {
            const float* chp = xg + (size_t)c * HW_;
            float v = w00 * chp[i00] + w01 * chp[i01]
                    + w10 * chp[i10] + w11 * chp[i11];
            smem[pl * SROW + (g * CG_ + c) * KPOS + k] = (_Float16)(v * m);
        }
    }
    __syncthreads();

    // ---- stage 2: K=576 GEMM on WMMA -------------------------------------
    int lane  = tid & 31;
    int wid   = tid >> 5;
    int wm    = wid & 1;
    int ntile = wid >> 1;
    int prow  = wm * 16 + (lane & 15);
    const _Float16* arow = smem + prow * SROW + ((lane >> 4) * 8);
    const _Float16* bp   = wdbuf + (size_t)ntile * 18 * 512 + lane * 16;

    v8f acc = {};
#pragma unroll
    for (int kc = 0; kc < 18; ++kc) {
        union { v4u u[2]; v16h h; } A, Bf;
        A.u[0]  = *(const v4u*)(arow + kc * 32);
        A.u[1]  = *(const v4u*)(arow + kc * 32 + 16);
        Bf.u[0] = *(const v4u*)(bp + kc * 512);
        Bf.u[1] = *(const v4u*)(bp + kc * 512 + 8);
        acc = __builtin_amdgcn_wmma_f32_16x16x32_f16(
            false, A.h, false, Bf.h, (short)0, acc, false, false);
    }

    int cout = ntile * 16 + (lane & 15);
    float bb = bias[cout];
    int pr0  = pb + wm * 16 + (lane >> 4) * 8;
    float* ob;
    if (n < 5)   // x_backward frames 0..4
        ob = out + ((size_t)n * C_ + cout) * HW_;
    else         // x_forward frames 1..5
        ob = out + (size_t)T_ * C_ * HW_ + ((size_t)(n - 4) * C_ + cout) * HW_;
#pragma unroll
    for (int r = 0; r < 8; ++r)
        ob[pr0 + r] = acc[r] + bb;
}

__global__ __launch_bounds__(256)
void fill_zero(float* __restrict__ p, int nelem)
{
    int i = blockIdx.x * blockDim.x + threadIdx.x;
    if (i < nelem) p[i] = 0.f;
}

// ---------------------------------------------------------------------------
extern "C" void kernel_launch(void* const* d_in, const int* in_sizes, int n_in,
                              void* d_out, int out_size, void* d_ws, size_t ws_size,
                              hipStream_t stream)
{
    const float* x     = (const float*)d_in[0];
    const float* flb   = (const float*)d_in[1];
    const float* flf   = (const float*)d_in[2];
    const float* cw1   = (const float*)d_in[3];
    const float* cb1   = (const float*)d_in[4];
    const float* cw2   = (const float*)d_in[5];
    const float* cb2   = (const float*)d_in[6];
    const float* cw3   = (const float*)d_in[7];
    const float* cb3   = (const float*)d_in[8];
    const float* cw4   = (const float*)d_in[9];
    const float* cb4   = (const float*)d_in[10];
    const float* dcw   = (const float*)d_in[11];
    const float* dcb   = (const float*)d_in[12];
    float* out = (float*)d_out;

    // workspace carve-up (all sizes are multiples of 256 B)
    char* ws = (char*)d_ws;
    size_t off = 0;
    _Float16* inp = (_Float16*)(ws + off); off += (size_t)NPAIR * HW_ * CIN1P * 2;
    _Float16* h1  = (_Float16*)(ws + off); off += (size_t)NPAIR * HW_ * C_ * 2;
    _Float16* h2  = (_Float16*)(ws + off); off += (size_t)NPAIR * HW_ * C_ * 2;
    _Float16* h3  = (_Float16*)(ws + off); off += (size_t)NPAIR * HW_ * C_ * 2;
    _Float16* c4  = (_Float16*)(ws + off); off += (size_t)NPAIR * HW_ * C4OUTP * 2;
    _Float16* w1b = (_Float16*)(ws + off); off += (size_t)(C_ / 16) * 9 * (CIN1P / 32) * 512 * 2;
    _Float16* w2b = (_Float16*)(ws + off); off += (size_t)(C_ / 16) * 9 * (C_ / 32) * 512 * 2;
    _Float16* w3b = (_Float16*)(ws + off); off += (size_t)(C_ / 16) * 9 * (C_ / 32) * 512 * 2;
    _Float16* w4b = (_Float16*)(ws + off); off += (size_t)(C4OUTP / 16) * 9 * (C_ / 32) * 512 * 2;
    _Float16* wdb = (_Float16*)(ws + off); off += (size_t)4 * 18 * 512 * 2;
    (void)off; (void)ws_size; (void)n_in; (void)in_sizes; (void)out_size;

    // --- weight pre-swizzle ---
    {
        long t1 = (long)(C_ / 16) * 9 * (CIN1P / 32) * 512;
        prep_conv_w<<<dim3((unsigned)((t1 + 255) / 256)), 256, 0, stream>>>(
            cw1, w1b, C_, C_, CIN1, CIN1P);
        long t2 = (long)(C_ / 16) * 9 * (C_ / 32) * 512;
        prep_conv_w<<<dim3((unsigned)((t2 + 255) / 256)), 256, 0, stream>>>(
            cw2, w2b, C_, C_, C_, C_);
        prep_conv_w<<<dim3((unsigned)((t2 + 255) / 256)), 256, 0, stream>>>(
            cw3, w3b, C_, C_, C_, C_);
        long t4 = (long)(C4OUTP / 16) * 9 * (C_ / 32) * 512;
        prep_conv_w<<<dim3((unsigned)((t4 + 255) / 256)), 256, 0, stream>>>(
            cw4, w4b, C4OUT, C4OUTP, C_, C_);
        int td = 4 * 18 * 512;
        prep_dcn_w<<<dim3((td + 255) / 256), 256, 0, stream>>>(dcw, wdb);
    }

    // --- flow warp + input assembly ---
    warp_build_inp<<<dim3((NPAIR * HW_ + 255) / 256), 256, 0, stream>>>(
        x, flb, flf, inp);

    // --- conv stack on WMMA (block tile: 128 px x 64 couts) ---
    dim3 blk(256);
    dim3 g1(HW_ / 128, 1, NPAIR);
    conv3x3_wmma<CIN1P, true><<<g1, blk, 0, stream>>>(inp, w1b, cb1, h1, C_, C_);
    conv3x3_wmma<C_,    true><<<g1, blk, 0, stream>>>(h1,  w2b, cb2, h2, C_, C_);
    conv3x3_wmma<C_,    true><<<g1, blk, 0, stream>>>(h2,  w3b, cb3, h3, C_, C_);
    dim3 g4(HW_ / 128, C4OUTP / 64, NPAIR);
    conv3x3_wmma<C_,   false><<<g4, blk, 0, stream>>>(h3,  w4b, cb4, c4, C4OUT, C4OUTP);

    // --- zero frames: x_backward[T-1], x_forward[0] ---
    int zf = C_ * HW_;
    fill_zero<<<dim3((zf + 255) / 256), 256, 0, stream>>>(
        out + (size_t)(T_ - 1) * C_ * HW_, zf);
    fill_zero<<<dim3((zf + 255) / 256), 256, 0, stream>>>(
        out + (size_t)T_ * C_ * HW_, zf);

    // --- modulated deformable conv ---
    dcn_wmma<<<dim3(HW_ / 32, 1, NPAIR), blk, 0, stream>>>(
        x, flb, flf, c4, wdb, dcb, out);
}
